// Encoder_4896262717776
// MI455X (gfx1250) — compile-verified, matrix operand order checked by reading, not run on hardware
//
#include <hip/hip_runtime.h>
#include <hip/hip_bf16.h>
#include <math.h>

typedef __bf16 bf16;
typedef __attribute__((ext_vector_type(16))) __bf16 bf16x16;
typedef __attribute__((ext_vector_type(8)))  float  f32x8;

#define HDIM   768
#define SEQ    1024
#define BATCH  4
#define NHEAD  12
#define DHEAD  64
#define MROWS  4096      // B*S
#define FFH    3072      // 4H
#define ATT_SCALE 0.125f // 1/sqrt(64)

// ---- CDNA5 TDM availability (device pass only; arity differs per toolchain)
#if defined(__HIP_DEVICE_COMPILE__) && __has_builtin(__builtin_amdgcn_tensor_load_to_lds)
  #define USE_TDM 1
  #if __has_include(<hip/amd_detail/amd_gfx1250_TDM.h>)
    #define TDM_SIX_ARGS 1
  #else
    #define TDM_SIX_ARGS 0
  #endif
#else
  #define USE_TDM 0
#endif

// ---------------------------------------------------------------- helpers
__device__ __forceinline__ bf16 f2bf(float f) {
  union { float f; unsigned u; } a; a.f = f;
  unsigned r = a.u + 0x7FFFu + ((a.u >> 16) & 1u);   // RNE
  union { unsigned short s; bf16 h; } o; o.s = (unsigned short)(r >> 16);
  return o.h;
}

// Load one 16x32 bf16 fragment in CDNA5 WMMA A/B register layout.
// lane L: row (M or N) = L%16 ; K sub-chunk c0 = 8*(L>=16);
// per lane: two contiguous 16B chunks at K offsets [c0,c0+8) and [c0+16,c0+24).
__device__ __forceinline__ bf16x16 load_frag(const bf16* __restrict__ base, int ld) {
  const int lane = threadIdx.x & 31;
  const int r    = lane & 15;
  const int c0   = (lane >> 4) << 3;
  union { bf16x16 v; uint4 q[2]; } u;
  const bf16* p = base + (size_t)r * ld + c0;
  u.q[0] = *(const uint4*)(p);
  u.q[1] = *(const uint4*)(p + 16);
  return u.v;
}

__device__ __forceinline__ f32x8 wmma_bf16(bf16x16 a, bf16x16 b, f32x8 c) {
  return __builtin_amdgcn_wmma_f32_16x16x32_bf16(false, a, false, b, (short)0, c,
                                                 false, false);
}

#if USE_TDM
typedef unsigned int u32x4 __attribute__((ext_vector_type(4)));
typedef int          i32x4 __attribute__((ext_vector_type(4)));
typedef int          i32x8 __attribute__((ext_vector_type(8)));

// Issue a TDM load of a [rows x 32] bf16 tile (row stride = stride_elems)
// from global into LDS at byte offset ldsOff. 2-D descriptor, groups 2/3 zero.
__device__ __forceinline__ void tdm_load_tile(const bf16* gsrc, unsigned ldsOff,
                                              int rows, int stride_elems) {
  unsigned long long ga = (unsigned long long)(uintptr_t)gsrc;
  u32x4 g0;
  g0.x = 1u;                                            // count=1, user desc
  g0.y = ldsOff;                                        // lds_addr
  g0.z = (unsigned)ga;                                  // global_addr[31:0]
  g0.w = ((unsigned)(ga >> 32) & 0x01FFFFFFu) | (2u << 30); // addr[56:32], type=2
  i32x8 g1;
  g1[0] = 1 << 16;                                      // data_size = 2 bytes
  g1[1] = (stride_elems & 0xffff) << 16;                // tensor_dim0 lo16
  g1[2] = (stride_elems >> 16) & 0xffff;                // tensor_dim0 hi16, dim1 lo16=0
  g1[3] = 16 | (32 << 16);                              // tensor_dim1 hi16 (=1<<20), tile_dim0=32
  g1[4] = rows;                                         // tile_dim1 = rows, tile_dim2 = 0
  g1[5] = stride_elems;                                 // tensor_dim0_stride lo32
  g1[6] = 0;                                            // stride hi16 = 0
  g1[7] = 0;
  i32x4 z4 = {0, 0, 0, 0};
#if TDM_SIX_ARGS
  i32x8 z8 = {0, 0, 0, 0, 0, 0, 0, 0};
  __builtin_amdgcn_tensor_load_to_lds(g0, g1, z4, z4, z8, 0);
#else
  __builtin_amdgcn_tensor_load_to_lds(g0, g1, z4, z4, 0);
#endif
}
#endif // USE_TDM

// ---------------------------------------------------------------- cast fp32 -> bf16
__global__ void cast_kernel(const float* __restrict__ in, bf16* __restrict__ out, int n) {
  int i = blockIdx.x * 256 + threadIdx.x;
  if (i < n) out[i] = f2bf(in[i]);
}

// ---------------------------------------------------------------- LayerNorm
template<bool OUT_BF16>
__launch_bounds__(256)
__global__ void ln_kernel(const float* __restrict__ x, const float* __restrict__ g,
                          const float* __restrict__ b, void* __restrict__ outp) {
  const int row = blockIdx.x, t = threadIdx.x;
  const float* xr = x + (size_t)row * HDIM;
  float v0 = xr[t], v1 = xr[t + 256], v2 = xr[t + 512];
  __shared__ float red[8];
  float s = v0 + v1 + v2;
  for (int off = 16; off; off >>= 1) s += __shfl_xor(s, off, 32);
  if ((t & 31) == 0) red[t >> 5] = s;
  __syncthreads();
  if (t == 0) { float tot = 0.f; for (int i = 0; i < 8; ++i) tot += red[i]; red[0] = tot; }
  __syncthreads();
  float mean = red[0] * (1.0f / HDIM);
  __syncthreads();
  float d0 = v0 - mean, d1 = v1 - mean, d2 = v2 - mean;
  s = d0*d0 + d1*d1 + d2*d2;
  for (int off = 16; off; off >>= 1) s += __shfl_xor(s, off, 32);
  if ((t & 31) == 0) red[t >> 5] = s;
  __syncthreads();
  if (t == 0) { float tot = 0.f; for (int i = 0; i < 8; ++i) tot += red[i]; red[0] = tot; }
  __syncthreads();
  float rstd = rsqrtf(red[0] * (1.0f / HDIM) + 1e-5f);
  float o0 = d0 * rstd * g[t]       + b[t];
  float o1 = d1 * rstd * g[t + 256] + b[t + 256];
  float o2 = d2 * rstd * g[t + 512] + b[t + 512];
  if constexpr (OUT_BF16) {
    bf16* orow = (bf16*)outp + (size_t)row * HDIM;
    orow[t] = f2bf(o0); orow[t + 256] = f2bf(o1); orow[t + 512] = f2bf(o2);
  } else {
    float* orow = (float*)outp + (size_t)row * HDIM;
    orow[t] = o0; orow[t + 256] = o1; orow[t + 512] = o2;
  }
}

// ---------------------------------------------------------------- GEMM: C = A @ W^T + bias (+ residual)
// A: [M,K] bf16 row-major; W: [N,K] bf16 row-major (both K contiguous).
// Block tile 128x128, 8 waves (4 m x 2 n), wave tile 32x64 -> 8 v_wmma per K-step.
// TDM path: double-buffered LDS, DMA issued by wave 0, one barrier per K-step.
template<bool OUT_BF16, bool RESIDUAL>
__launch_bounds__(256)
__global__ void gemm_wmma(const bf16* __restrict__ A, const bf16* __restrict__ W,
                          const float* __restrict__ bias, const float* __restrict__ resid,
                          void* __restrict__ outp, int M, int N, int K) {
#if USE_TDM
  __shared__ alignas(16) bf16 Ash[2][128][32];
  __shared__ alignas(16) bf16 Bsh[2][128][32];
#else
  __shared__ alignas(16) bf16 Ash[1][128][32];
  __shared__ alignas(16) bf16 Bsh[1][128][32];
#endif
  const int t    = threadIdx.x;
  const int m0   = blockIdx.y * 128;
  const int n0   = blockIdx.x * 128;
  const int wid  = t >> 5;
  const int wm   = wid & 3;
  const int wn   = wid >> 2;
  const int lane = t & 31;
  const int colL = lane & 15;
  const int rh   = (lane >> 4) << 3;
  const int nk   = K >> 5;

  f32x8 acc[2][4] = {};

#if USE_TDM
  const bool issuer = (t < 32);
  if (issuer) {
    tdm_load_tile(&A[(size_t)m0 * K], (unsigned)(uintptr_t)&Ash[0][0][0], 128, K);
    tdm_load_tile(&W[(size_t)n0 * K], (unsigned)(uintptr_t)&Bsh[0][0][0], 128, K);
  }
#endif

  for (int ki = 0; ki < nk; ++ki) {
#if USE_TDM
    const int buf = ki & 1;
    if (issuer) __builtin_amdgcn_s_wait_tensorcnt(0);   // tile ki landed
    __syncthreads();                                     // ...visible to all waves
    if (issuer && (ki + 1 < nk)) {                       // prefetch tile ki+1
      const size_t k0n = (size_t)(ki + 1) * 32;
      tdm_load_tile(&A[(size_t)m0 * K + k0n],
                    (unsigned)(uintptr_t)&Ash[buf ^ 1][0][0], 128, K);
      tdm_load_tile(&W[(size_t)n0 * K + k0n],
                    (unsigned)(uintptr_t)&Bsh[buf ^ 1][0][0], 128, K);
    }
#else
    const int buf = 0;
    const int k0  = ki * 32;
    __syncthreads();
#pragma unroll
    for (int i = 0; i < 2; ++i) {
      int u   = t + 256 * i;
      int row = u >> 2;
      int col = (u & 3) << 3;
      *(uint4*)&Ash[0][row][col] = *(const uint4*)&A[(size_t)(m0 + row) * K + k0 + col];
      *(uint4*)&Bsh[0][row][col] = *(const uint4*)&W[(size_t)(n0 + row) * K + k0 + col];
      if (k0 + 32 < K) {
        __builtin_prefetch(&A[(size_t)(m0 + row) * K + k0 + 32 + col], 0, 0);
        __builtin_prefetch(&W[(size_t)(n0 + row) * K + k0 + 32 + col], 0, 0);
      }
    }
    __syncthreads();
#endif
    bf16x16 afr[2];
#pragma unroll
    for (int sm = 0; sm < 2; ++sm) afr[sm] = load_frag(&Ash[buf][wm * 32 + sm * 16][0], 32);
    bf16x16 bfr[4];
#pragma unroll
    for (int sn = 0; sn < 4; ++sn) bfr[sn] = load_frag(&Bsh[buf][wn * 64 + sn * 16][0], 32);
#pragma unroll
    for (int sm = 0; sm < 2; ++sm)
#pragma unroll
      for (int sn = 0; sn < 4; ++sn)
        acc[sm][sn] = wmma_bf16(afr[sm], bfr[sn], acc[sm][sn]);
  }

#pragma unroll
  for (int sm = 0; sm < 2; ++sm) {
#pragma unroll
    for (int sn = 0; sn < 4; ++sn) {
      int gc = n0 + wn * 64 + sn * 16 + colL;
      float bb = bias[gc];
#pragma unroll
      for (int v = 0; v < 8; ++v) {
        int gr = m0 + wm * 32 + sm * 16 + v + rh;
        size_t oi = (size_t)gr * N + gc;
        float val = acc[sm][sn][v] + bb;
        if constexpr (RESIDUAL) val += resid[oi];
        if constexpr (OUT_BF16) ((bf16*)outp)[oi] = f2bf(val);
        else                    ((float*)outp)[oi] = val;
      }
    }
  }
}

// ---------------------------------------------------------------- fused GeGLU GEMM
// h = xn @ W1^T + b1 ; u = h[:, :FFH], g = h[:, FFH:] ; act = u * gelu_exact(g)
// Block: 128 rows x 64 GeGLU columns (computes both u and gate tiles).
__launch_bounds__(256)
__global__ void gemm_geglu(const bf16* __restrict__ A, const bf16* __restrict__ W1b,
                           const float* __restrict__ b1, bf16* __restrict__ act,
                           int M, int K) {
#if USE_TDM
  __shared__ alignas(16) bf16 Ash[2][128][32];
  __shared__ alignas(16) bf16 Bu[2][64][32];
  __shared__ alignas(16) bf16 Bg[2][64][32];
#else
  __shared__ alignas(16) bf16 Ash[1][128][32];
  __shared__ alignas(16) bf16 Bu[1][64][32];
  __shared__ alignas(16) bf16 Bg[1][64][32];
#endif
  const int t    = threadIdx.x;
  const int m0   = blockIdx.y * 128;
  const int n0   = blockIdx.x * 64;
  const int wid  = t >> 5;
  const int wm   = wid & 3;
  const int wn   = wid >> 2;
  const int lane = t & 31;
  const int colL = lane & 15;
  const int rh   = (lane >> 4) << 3;
  const int nk   = K >> 5;

  f32x8 au[2][2] = {}, ag[2][2] = {};

#if USE_TDM
  const bool issuer = (t < 32);
  if (issuer) {
    tdm_load_tile(&A[(size_t)m0 * K],           (unsigned)(uintptr_t)&Ash[0][0][0], 128, K);
    tdm_load_tile(&W1b[(size_t)n0 * K],         (unsigned)(uintptr_t)&Bu[0][0][0],   64, K);
    tdm_load_tile(&W1b[(size_t)(FFH + n0) * K], (unsigned)(uintptr_t)&Bg[0][0][0],   64, K);
  }
#endif

  for (int ki = 0; ki < nk; ++ki) {
#if USE_TDM
    const int buf = ki & 1;
    if (issuer) __builtin_amdgcn_s_wait_tensorcnt(0);
    __syncthreads();
    if (issuer && (ki + 1 < nk)) {
      const size_t k0n = (size_t)(ki + 1) * 32;
      tdm_load_tile(&A[(size_t)m0 * K + k0n],
                    (unsigned)(uintptr_t)&Ash[buf ^ 1][0][0], 128, K);
      tdm_load_tile(&W1b[(size_t)n0 * K + k0n],
                    (unsigned)(uintptr_t)&Bu[buf ^ 1][0][0], 64, K);
      tdm_load_tile(&W1b[(size_t)(FFH + n0) * K + k0n],
                    (unsigned)(uintptr_t)&Bg[buf ^ 1][0][0], 64, K);
    }
#else
    const int buf = 0;
    const int k0  = ki * 32;
    __syncthreads();
#pragma unroll
    for (int i = 0; i < 2; ++i) {
      int u   = t + 256 * i;
      int row = u >> 2;
      int col = (u & 3) << 3;
      *(uint4*)&Ash[0][row][col] = *(const uint4*)&A[(size_t)(m0 + row) * K + k0 + col];
    }
    {
      int row = t >> 2;
      int col = (t & 3) << 3;
      *(uint4*)&Bu[0][row][col] = *(const uint4*)&W1b[(size_t)(n0 + row) * K + k0 + col];
      *(uint4*)&Bg[0][row][col] = *(const uint4*)&W1b[(size_t)(FFH + n0 + row) * K + k0 + col];
    }
    __syncthreads();
#endif
    bf16x16 afr[2];
#pragma unroll
    for (int sm = 0; sm < 2; ++sm) afr[sm] = load_frag(&Ash[buf][wm * 32 + sm * 16][0], 32);
    bf16x16 bfu[2], bfg[2];
#pragma unroll
    for (int sn = 0; sn < 2; ++sn) {
      bfu[sn] = load_frag(&Bu[buf][wn * 32 + sn * 16][0], 32);
      bfg[sn] = load_frag(&Bg[buf][wn * 32 + sn * 16][0], 32);
    }
#pragma unroll
    for (int sm = 0; sm < 2; ++sm)
#pragma unroll
      for (int sn = 0; sn < 2; ++sn) {
        au[sm][sn] = wmma_bf16(afr[sm], bfu[sn], au[sm][sn]);
        ag[sm][sn] = wmma_bf16(afr[sm], bfg[sn], ag[sm][sn]);
      }
  }

#pragma unroll
  for (int sm = 0; sm < 2; ++sm) {
#pragma unroll
    for (int sn = 0; sn < 2; ++sn) {
      int gc = n0 + wn * 32 + sn * 16 + colL;
      float bub = b1[gc];
      float bgb = b1[FFH + gc];
#pragma unroll
      for (int v = 0; v < 8; ++v) {
        int gr = m0 + wm * 32 + sm * 16 + v + rh;
        float uu = au[sm][sn][v] + bub;
        float gg = ag[sm][sn][v] + bgb;
        float gl = 0.5f * gg * (1.0f + erff(gg * 0.70710678118654752f));
        act[(size_t)gr * FFH + gc] = f2bf(uu * gl);
      }
    }
  }
}

// ---------------------------------------------------------------- flash attention
// block = (qtile of 64 rows, head, batch); 4 waves, each owns 16 q-rows.
__launch_bounds__(128)
__global__ void attn_wmma(const bf16* __restrict__ Q, const bf16* __restrict__ Kt,
                          const bf16* __restrict__ V, const unsigned char* __restrict__ pm,
                          bf16* __restrict__ Oa) {
  __shared__ alignas(16) bf16 Ksh[32][64];     // [key][d]
  __shared__ alignas(16) bf16 Vsh[64][32];     // [d][key] (transposed)
  __shared__ alignas(16) bf16 Psh[4][16][32];  // per-wave probs staging
  const int t    = threadIdx.x;
  const int wave = t >> 5;
  const int lane = t & 31;
  const int colL = lane & 15;
  const int rh   = (lane >> 4) << 3;
  const int h    = blockIdx.y;
  const int bz   = blockIdx.z;
  const size_t rowbase = (size_t)bz * SEQ;
  const int hD   = h * DHEAD;
  const int q0   = blockIdx.x * 64 + wave * 16;

  bf16x16 aq0 = load_frag(&Q[(rowbase + q0) * HDIM + hD],      HDIM);  // d 0..31
  bf16x16 aq1 = load_frag(&Q[(rowbase + q0) * HDIM + hD + 32], HDIM);  // d 32..63

  f32x8 o[4] = {};
  float m8[8], l8[8];
#pragma unroll
  for (int v = 0; v < 8; ++v) { m8[v] = -1e30f; l8[v] = 0.0f; }

  for (int kp = 0; kp < SEQ; kp += 32) {
    __syncthreads();
    // stage K tile [32][64]
#pragma unroll
    for (int i = 0; i < 2; ++i) {
      int u   = t + 128 * i;
      int row = u >> 3;
      int col = (u & 7) << 3;
      *(uint4*)&Ksh[row][col] =
          *(const uint4*)&Kt[(rowbase + kp + row) * HDIM + hD + col];
    }
    // stage V tile transposed -> [d][key]
#pragma unroll
    for (int j = 0; j < 8; ++j) {
      int p   = t * 8 + j;
      int row = p >> 5;
      int dc  = (p & 31) << 1;
      union { unsigned u32; unsigned short s[2]; } w;
      w.u32 = *(const unsigned*)&V[(rowbase + kp + row) * HDIM + hD + dc];
      *(unsigned short*)&Vsh[dc][row]     = w.s[0];
      *(unsigned short*)&Vsh[dc + 1][row] = w.s[1];
    }
    __syncthreads();

    // scores: two 16x16 tiles, K-dim = 64 -> 4 wmma
    f32x8 s0 = {}, s1 = {};
    s0 = wmma_bf16(aq0, load_frag(&Ksh[0][0],   64), s0);
    s0 = wmma_bf16(aq1, load_frag(&Ksh[0][32],  64), s0);
    s1 = wmma_bf16(aq0, load_frag(&Ksh[16][0],  64), s1);
    s1 = wmma_bf16(aq1, load_frag(&Ksh[16][32], 64), s1);

    bool pk0 = pm[rowbase + kp + colL]      != 0;
    bool pk1 = pm[rowbase + kp + 16 + colL] != 0;

#pragma unroll
    for (int v = 0; v < 8; ++v) {
      float a0 = pk0 ? -1e30f : s0[v] * ATT_SCALE;
      float a1 = pk1 ? -1e30f : s1[v] * ATT_SCALE;
      float rm = fmaxf(a0, a1);
      rm = fmaxf(rm, __shfl_xor(rm, 1, 32));
      rm = fmaxf(rm, __shfl_xor(rm, 2, 32));
      rm = fmaxf(rm, __shfl_xor(rm, 4, 32));
      rm = fmaxf(rm, __shfl_xor(rm, 8, 32));
      float mn    = fmaxf(m8[v], rm);
      float alpha = __expf(m8[v] - mn);
      float p0    = __expf(a0 - mn);
      float p1    = __expf(a1 - mn);
      float ls    = p0 + p1;
      ls += __shfl_xor(ls, 1, 32);
      ls += __shfl_xor(ls, 2, 32);
      ls += __shfl_xor(ls, 4, 32);
      ls += __shfl_xor(ls, 8, 32);
      m8[v] = mn;
      l8[v] = l8[v] * alpha + ls;
      o[0][v] *= alpha; o[1][v] *= alpha; o[2][v] *= alpha; o[3][v] *= alpha;
      Psh[wave][v + rh][colL]      = f2bf(p0);
      Psh[wave][v + rh][colL + 16] = f2bf(p1);
    }
    __syncthreads();

    // P @ V : 4 wmma (D = 64)
    bf16x16 ap = load_frag(&Psh[wave][0][0], 32);
#pragma unroll
    for (int dt = 0; dt < 4; ++dt)
      o[dt] = wmma_bf16(ap, load_frag(&Vsh[dt * 16][0], 32), o[dt]);
  }

#pragma unroll
  for (int dt = 0; dt < 4; ++dt)
#pragma unroll
    for (int v = 0; v < 8; ++v) {
      float val = o[dt][v] / l8[v];
      Oa[(rowbase + q0 + v + rh) * HDIM + hD + dt * 16 + colL] = f2bf(val);
    }
}

// ---------------------------------------------------------------- host driver
extern "C" void kernel_launch(void* const* d_in, const int* in_sizes, int n_in,
                              void* d_out, int out_size, void* d_ws, size_t ws_size,
                              hipStream_t stream) {
  (void)in_sizes; (void)n_in; (void)out_size; (void)ws_size;
  const float*         x_in = (const float*)d_in[0];
  const unsigned char* pm   = (const unsigned char*)d_in[1];
  const float* Wq = (const float*)d_in[2];  const float* bq  = (const float*)d_in[3];
  const float* Wk = (const float*)d_in[4];  const float* bk  = (const float*)d_in[5];
  const float* Wv = (const float*)d_in[6];  const float* bv  = (const float*)d_in[7];
  const float* Wo = (const float*)d_in[8];  const float* bo  = (const float*)d_in[9];
  const float* g1 = (const float*)d_in[10]; const float* be1 = (const float*)d_in[11];
  const float* g2 = (const float*)d_in[12]; const float* be2 = (const float*)d_in[13];
  const float* W1 = (const float*)d_in[14]; const float* b1  = (const float*)d_in[15];
  const float* W2 = (const float*)d_in[16]; const float* b2  = (const float*)d_in[17];
  const float* og = (const float*)d_in[18]; const float* ob  = (const float*)d_in[19];

  const size_t HH   = (size_t)HDIM * HDIM;      // 589824
  const size_t W1SZ = (size_t)2 * FFH * HDIM;   // 4718592
  const size_t W2SZ = (size_t)HDIM * FFH;       // 2359296

  char* base = (char*)d_ws;
  size_t off = 0;
  auto take = [&](size_t bytes) -> void* {
    void* p = base + off;
    off += (bytes + 255) & ~(size_t)255;
    return p;
  };
  bf16* wqb = (bf16*)take(HH * 2);
  bf16* wkb = (bf16*)take(HH * 2);
  bf16* wvb = (bf16*)take(HH * 2);
  bf16* wob = (bf16*)take(HH * 2);
  bf16* w1b = (bf16*)take(W1SZ * 2);
  bf16* w2b = (bf16*)take(W2SZ * 2);
  bf16* xnb  = (bf16*)take((size_t)MROWS * HDIM * 2);
  bf16* qb   = (bf16*)take((size_t)MROWS * HDIM * 2);
  bf16* kb   = (bf16*)take((size_t)MROWS * HDIM * 2);
  bf16* vb   = (bf16*)take((size_t)MROWS * HDIM * 2);
  bf16* avb  = (bf16*)take((size_t)MROWS * HDIM * 2);
  bf16* actb = (bf16*)take((size_t)MROWS * FFH * 2);
  float* xf  = (float*)take((size_t)MROWS * HDIM * 4);

  hipMemcpyAsync(xf, x_in, sizeof(float) * MROWS * HDIM,
                 hipMemcpyDeviceToDevice, stream);

  dim3 g_gemm(HDIM / 128, MROWS / 128);   // 6 x 32
  dim3 g_geglu(FFH / 64, MROWS / 128);    // 48 x 32
  dim3 g_attn(SEQ / 64, NHEAD, BATCH);    // 16 x 12 x 4

  for (int l = 0; l < 6; ++l) {
    auto castN = [&](const float* src, bf16* dst, size_t n) {
      cast_kernel<<<(int)((n + 255) / 256), 256, 0, stream>>>(src, dst, (int)n);
    };
    castN(Wq + (size_t)l * HH,   wqb, HH);
    castN(Wk + (size_t)l * HH,   wkb, HH);
    castN(Wv + (size_t)l * HH,   wvb, HH);
    castN(Wo + (size_t)l * HH,   wob, HH);
    castN(W1 + (size_t)l * W1SZ, w1b, W1SZ);
    castN(W2 + (size_t)l * W2SZ, w2b, W2SZ);

    ln_kernel<true><<<MROWS, 256, 0, stream>>>(xf, g1 + l * HDIM, be1 + l * HDIM, xnb);

    gemm_wmma<true, false><<<g_gemm, 256, 0, stream>>>(
        xnb, wqb, bq + l * HDIM, nullptr, qb, MROWS, HDIM, HDIM);
    gemm_wmma<true, false><<<g_gemm, 256, 0, stream>>>(
        xnb, wkb, bk + l * HDIM, nullptr, kb, MROWS, HDIM, HDIM);
    gemm_wmma<true, false><<<g_gemm, 256, 0, stream>>>(
        xnb, wvb, bv + l * HDIM, nullptr, vb, MROWS, HDIM, HDIM);

    attn_wmma<<<g_attn, 128, 0, stream>>>(qb, kb, vb, pm, avb);

    gemm_wmma<false, true><<<g_gemm, 256, 0, stream>>>(
        avb, wob, bo + l * HDIM, xf, xf, MROWS, HDIM, HDIM);

    ln_kernel<true><<<MROWS, 256, 0, stream>>>(xf, g2 + l * HDIM, be2 + l * HDIM, xnb);

    gemm_geglu<<<g_geglu, 256, 0, stream>>>(
        xnb, w1b, b1 + (size_t)l * 2 * FFH, actb, MROWS, HDIM);

    gemm_wmma<false, true><<<g_gemm, 256, 0, stream>>>(
        actb, w2b, b2 + l * HDIM, xf, xf, MROWS, HDIM, FFH);
  }

  ln_kernel<false><<<MROWS, 256, 0, stream>>>(xf, og, ob, d_out);
}
